// pointnet2_seg_2_76175539962307
// MI455X (gfx1250) — matrix operand order repaired and checked by reading, not compiled, not run on hardware
//
#include <hip/hip_runtime.h>
#include <hip/hip_bf16.h>
#include <math.h>

// ---------------------------------------------------------------------------
// PointNet++ MSG segmentation forward for MI455X (gfx1250, wave32, WMMA).
// All channel-mixing MLP layers run on v_wmma_f32_16x16x32_f16 (f16 in, f32 acc).
// Weights are pre-converted once per call into padded f16 (coutp16 x kpad32)
// so WMMA fragment loads are unconditional contiguous dword loads.
// ---------------------------------------------------------------------------

typedef __attribute__((ext_vector_type(16))) _Float16 v16h;
typedef __attribute__((ext_vector_type(8)))  float    v8f;
typedef __attribute__((ext_vector_type(2)))  _Float16 h2;

static constexpr int kB  = 16;     // batch
static constexpr int kN  = 4096;   // input points

struct LayerP {
  const float *W, *b, *gamma, *beta, *mean, *var;
};

__host__ __device__ constexpr int rup16(int x) { return (x + 15) & ~15; }
__host__ __device__ constexpr int rup32(int x) { return (x + 31) & ~31; }

// ---------------------------------------------------------------------------
// WMMA helpers (CDNA5 VGPR layouts, wave32)
// ---------------------------------------------------------------------------
__device__ __forceinline__ v8f wmma_f16(v16h a, v16h b, v8f c) {
  return __builtin_amdgcn_wmma_f32_16x16x32_f16(
      /*neg_a=*/false, a, /*neg_b=*/false, b,
      /*c_mod=*/(short)0, c, /*reuse_a=*/false, /*reuse_b=*/false);
}

// A fragment: 16x32 f16, row-major source with row stride lda (elements).
// lane L: M = L%16, half = L/16; vgpr pair e: K = (e<4 ? 2e : 16+2(e-4)) + 8*half
__device__ __forceinline__ v16h load_a_frag(const _Float16* A, int lda, int m0, int k0) {
  int lane = threadIdx.x & 31;
  int m = m0 + (lane & 15);
  int half = lane >> 4;
  const _Float16* r = A + (size_t)m * lda + k0 + 8 * half;
  v16h a;
#pragma unroll
  for (int e = 0; e < 8; ++e) {
    int kb = (e < 4) ? (2 * e) : (8 + 2 * e);   // (e<4?2e:16+2(e-4)) rebased by -8*half
    h2 p = *(const h2*)(r + kb);
    a[2 * e]     = p.x;
    a[2 * e + 1] = p.y;
  }
  return a;
}

// B fragment: 32x16, B[k][n] = Wh[n][k]; Wh is (coutp16 x ldw) f16, zero-padded.
// lane L: N = L%16, half = L/16; pair e: K = (e<4 ? 2e : 16+2(e-4)) + 8*half
__device__ __forceinline__ v16h load_b_frag_h(const _Float16* Wh, int ldw,
                                              int n0, int k0) {
  int lane = threadIdx.x & 31;
  int n = n0 + (lane & 15);
  int half = lane >> 4;
  const _Float16* r = Wh + (size_t)n * ldw + k0 + 8 * half;
  v16h b;
#pragma unroll
  for (int e = 0; e < 8; ++e) {
    h2 p = *(const h2*)(r + ((e < 4) ? (2 * e) : (8 + 2 * e)));
    b[2 * e]     = p.x;
    b[2 * e + 1] = p.y;
  }
  return b;
}

// ---------------------------------------------------------------------------
// Weight conversion: W f32 (cout x cin) -> Wh f16 (coutp x kpad), zero padded.
// ---------------------------------------------------------------------------
__global__ void convert_w_kernel(const float* __restrict__ W, int cout, int cin,
                                 _Float16* __restrict__ Wh, int coutp, int kpad) {
  int g = blockIdx.x * blockDim.x + threadIdx.x;
  int total = coutp * kpad;
  if (g >= total) return;
  int n = g / kpad, k = g % kpad;
  float v = (n < cout && k < cin) ? W[(size_t)n * cin + k] : 0.f;
  Wh[g] = (_Float16)v;
}

// ---------------------------------------------------------------------------
// Farthest point sampling: one block per batch, dmin in LDS, block argmax.
// ---------------------------------------------------------------------------
template <int N_IN, int NPOINT>
__global__ void fps_kernel(const float* __restrict__ xyz,
                           int* __restrict__ idx_out,
                           float* __restrict__ new_xyz) {
  __shared__ float dmin[N_IN];
  __shared__ float red_v[256];
  __shared__ int   red_i[256];
  const int b = blockIdx.x;
  const int t = threadIdx.x;
  const float* X = xyz + (size_t)b * N_IN * 3;
  for (int i = t; i < N_IN; i += 256) dmin[i] = 1e10f;
  __syncthreads();
  int far = 0;
  for (int s = 0; s < NPOINT; ++s) {
    if (t == 0) {
      idx_out[b * NPOINT + s] = far;
      new_xyz[((size_t)b * NPOINT + s) * 3 + 0] = X[far * 3 + 0];
      new_xyz[((size_t)b * NPOINT + s) * 3 + 1] = X[far * 3 + 1];
      new_xyz[((size_t)b * NPOINT + s) * 3 + 2] = X[far * 3 + 2];
    }
    float cx = X[far * 3 + 0], cy = X[far * 3 + 1], cz = X[far * 3 + 2];
    float best = -1.f; int bi = 0;
    for (int i = t; i < N_IN; i += 256) {
      float dx = X[i * 3 + 0] - cx;
      float dy = X[i * 3 + 1] - cy;
      float dz = X[i * 3 + 2] - cz;
      float d = dx * dx + dy * dy + dz * dz;
      float m = fminf(dmin[i], d);
      dmin[i] = m;
      if (m > best) { best = m; bi = i; }
    }
    red_v[t] = best; red_i[t] = bi;
    __syncthreads();
    for (int o = 128; o > 0; o >>= 1) {
      if (t < o) {
        if (red_v[t + o] > red_v[t] ||
            (red_v[t + o] == red_v[t] && red_i[t + o] < red_i[t])) {
          red_v[t] = red_v[t + o]; red_i[t] = red_i[t + o];
        }
      }
      __syncthreads();
    }
    far = red_i[0];
    __syncthreads();
  }
}

// ---------------------------------------------------------------------------
// Ball query: first K in-radius indices (ascending), pad with first hit.
// ---------------------------------------------------------------------------
__global__ void ball_query_kernel(const float* __restrict__ xyz, int n,
                                  const float* __restrict__ new_xyz, int total,
                                  float r2, int K, int* __restrict__ idx) {
  int g = blockIdx.x * blockDim.x + threadIdx.x;
  if (g >= total) return;
  int b = g / (total / kB);
  const float* X = xyz + (size_t)b * n * 3;
  float cx = new_xyz[g * 3 + 0], cy = new_xyz[g * 3 + 1], cz = new_xyz[g * 3 + 2];
  int* out = idx + (size_t)g * K;
  int cnt = 0, first = -1;
  for (int i = 0; i < n && cnt < K; ++i) {
    float dx = X[i * 3 + 0] - cx;
    float dy = X[i * 3 + 1] - cy;
    float dz = X[i * 3 + 2] - cz;
    float d = dx * dx + dy * dy + dz * dz;
    if (d <= r2) {
      if (first < 0) first = i;
      out[cnt++] = i;
    }
  }
  if (first < 0) first = 0;
  for (; cnt < K; ++cnt) out[cnt] = first;
}

// ---------------------------------------------------------------------------
// In-block GEMM layer (LDS -> LDS), 8 waves share the 16x16 tile grid.
// ---------------------------------------------------------------------------
template <int KS>
__device__ void block_layer(const _Float16* src, int sstride, int kpad,
                            const _Float16* Wh, int ldw,
                            _Float16* dst, int dstride, int cout, const LayerP& L) {
  const int coutp = rup32(cout);
  const int padw = coutp - cout;
  for (int e = threadIdx.x; e < KS * padw; e += blockDim.x) {
    int row = e / padw;
    int c = cout + (e % padw);
    dst[row * dstride + c] = (_Float16)0.f;
  }
  const int wave = threadIdx.x >> 5;
  const int nwaves = blockDim.x >> 5;
  const int mt = KS / 16;
  const int nt = (cout + 15) >> 4;
  for (int tile = wave; tile < mt * nt; tile += nwaves) {
    int mi = tile % mt, ni = tile / mt;
    v8f acc = {};
    for (int k0 = 0; k0 < kpad; k0 += 32) {
      v16h a = load_a_frag(src, sstride, mi * 16, k0);
      v16h bm = load_b_frag_h(Wh, ldw, ni * 16, k0);
      acc = wmma_f16(a, bm, acc);
    }
    int lane = threadIdx.x & 31;
    int n = ni * 16 + (lane & 15);
    int half = lane >> 4;
    if (n < cout) {
      float sc = L.gamma[n] * rsqrtf(L.var[n] + 1e-5f);
      float sh = (L.b[n] - L.mean[n]) * sc + L.beta[n];
#pragma unroll
      for (int v = 0; v < 8; ++v) {
        int m = mi * 16 + v + 8 * half;
        float y = fmaxf(acc[v] * sc + sh, 0.f);
        dst[m * dstride + n] = (_Float16)y;
      }
    }
  }
}

// ---------------------------------------------------------------------------
// Fused SA-MSG group kernel: one block = one (b,s) group.
// Gather (xyz-center | feats) -> 3 WMMA MLP layers in LDS -> max over KS ->
// write f32 slice of (B, ctot, S).
// ---------------------------------------------------------------------------
template <int KS, int CIN_RAW, int CIN_PAD, int C1, int C2, int C3, int B0C, int B1C>
__global__ void sa_group_mlp_kernel(
    const float* __restrict__ xyz, int npts_in,
    const float* __restrict__ feats, int cf, int feats_chmajor,
    const int* __restrict__ idx, const float* __restrict__ new_xyz, int S,
    const _Float16* __restrict__ Wh1, const _Float16* __restrict__ Wh2,
    const _Float16* __restrict__ Wh3,
    LayerP L1, LayerP L2, LayerP L3,
    float* __restrict__ outF, int ctot, int coff) {
  extern __shared__ _Float16 smem[];
  _Float16* buf0 = smem;
  _Float16* buf1 = smem + KS * B0C;

  const int bs = blockIdx.x;
  const int b = bs / S, s = bs % S;
  const int* gi = idx + (size_t)bs * KS;
  const float cx = new_xyz[bs * 3 + 0];
  const float cy = new_xyz[bs * 3 + 1];
  const float cz = new_xyz[bs * 3 + 2];

  for (int e = threadIdx.x; e < KS * CIN_PAD; e += blockDim.x) {
    int kk = e / CIN_PAD, c = e % CIN_PAD;
    int p = gi[kk];
    float v = 0.f;
    if (c < 3) {
      float ctr = (c == 0) ? cx : (c == 1) ? cy : cz;
      v = xyz[((size_t)b * npts_in + p) * 3 + c] - ctr;
    } else if (c < CIN_RAW) {
      if (feats_chmajor)
        v = feats[((size_t)b * cf + (c - 3)) * npts_in + p];
      else
        v = xyz[((size_t)b * npts_in + p) * 3 + (c - 3)];
    }
    buf0[kk * B0C + c] = (_Float16)v;
  }
  __syncthreads();
  block_layer<KS>(buf0, B0C, CIN_PAD, Wh1, CIN_PAD, buf1, B1C, C1, L1);
  __syncthreads();
  block_layer<KS>(buf1, B1C, rup32(C1), Wh2, rup32(C1), buf0, B0C, C2, L2);
  __syncthreads();
  block_layer<KS>(buf0, B0C, rup32(C2), Wh3, rup32(C2), buf1, B1C, C3, L3);
  __syncthreads();
  for (int c = threadIdx.x; c < C3; c += blockDim.x) {
    float mx = 0.f;  // outputs are post-ReLU
    for (int kk = 0; kk < KS; ++kk)
      mx = fmaxf(mx, (float)buf1[kk * B1C + c]);
    outF[((size_t)b * ctot + (coff + c)) * S + s] = mx;
  }
}

// ---------------------------------------------------------------------------
// Global GEMM + BN/ReLU: out[m][n] = act(A[m][:] * Wh[n][:] + b[n]).
// One wave per block, one 16x16 tile. M, cout multiples of 16.
// flags bit0: BN+ReLU (else bias only).
// ---------------------------------------------------------------------------
__global__ void gemm_bn_kernel(const _Float16* __restrict__ A, int lda,
                               const _Float16* __restrict__ Wh, int kpad,
                               LayerP L, int cout,
                               _Float16* __restrict__ outH, int ldo,
                               float* __restrict__ outF, int npts, int flags) {
  const int mi = blockIdx.x, ni = blockIdx.y;
  v8f acc = {};
  for (int k0 = 0; k0 < kpad; k0 += 32) {
    v16h a = load_a_frag(A, lda, mi * 16, k0);
    v16h bm = load_b_frag_h(Wh, kpad, ni * 16, k0);
    acc = wmma_f16(a, bm, acc);
  }
  const int lane = threadIdx.x & 31;
  const int n = ni * 16 + (lane & 15);
  const int half = lane >> 4;
  float sc = 1.f, sh = L.b[n];
  if (flags & 1) {
    sc = L.gamma[n] * rsqrtf(L.var[n] + 1e-5f);
    sh = (L.b[n] - L.mean[n]) * sc + L.beta[n];
  }
#pragma unroll
  for (int v = 0; v < 8; ++v) {
    int m = mi * 16 + v + 8 * half;
    float y = acc[v] * sc + sh;
    if (flags & 1) y = fmaxf(y, 0.f);
    if (outH) outH[(size_t)m * ldo + n] = (_Float16)y;
    if (outF) {
      int b = m / npts, p = m % npts;
      outF[((size_t)b * cout + n) * npts + p] = y;
    }
  }
}

// ---------------------------------------------------------------------------
// Builders / pooling / NN / epilogue kernels
// ---------------------------------------------------------------------------
__global__ void build_sa3_kernel(const float* __restrict__ l2_xyz,
                                 const float* __restrict__ l2p,
                                 _Float16* __restrict__ A, int total) {
  int e = blockIdx.x * blockDim.x + threadIdx.x;
  if (e >= total) return;
  int c = e % 544, bp = e / 544;
  int p = bp % 128, b = bp / 128;
  float v = 0.f;
  if (c < 3)        v = l2_xyz[((size_t)b * 128 + p) * 3 + c];
  else if (c < 515) v = l2p[((size_t)b * 512 + (c - 3)) * 128 + p];
  A[e] = (_Float16)v;
}

__global__ void maxpool_rows_kernel(const _Float16* __restrict__ A, int rows,
                                    int cols, float* __restrict__ out, int total) {
  int g = blockIdx.x * blockDim.x + threadIdx.x;
  if (g >= total) return;
  int b = g / cols, c = g % cols;
  const _Float16* base = A + (size_t)b * rows * cols + c;
  float mx = 0.f;
  for (int r = 0; r < rows; ++r) mx = fmaxf(mx, (float)base[(size_t)r * cols]);
  out[g] = mx;
}

__global__ void build_fp3_kernel(const float* __restrict__ l3,
                                 const float* __restrict__ l2p,
                                 _Float16* __restrict__ A, int total) {
  int e = blockIdx.x * blockDim.x + threadIdx.x;
  if (e >= total) return;
  int c = e % 1536, bp = e / 1536;
  int p = bp % 128, b = bp / 128;
  float v;
  if (c < 1024) v = l3[(size_t)b * 1024 + c];
  else          v = l2p[((size_t)b * 512 + (c - 1024)) * 128 + p];
  A[e] = (_Float16)v;
}

__global__ void three_nn_kernel(const float* __restrict__ unk, int nu,
                                const float* __restrict__ kn, int nk,
                                int* __restrict__ idx, float* __restrict__ w,
                                int total) {
  int g = blockIdx.x * blockDim.x + threadIdx.x;
  if (g >= total) return;
  int b = g / nu;
  float ux = unk[g * 3 + 0], uy = unk[g * 3 + 1], uz = unk[g * 3 + 2];
  const float* K = kn + (size_t)b * nk * 3;
  float d0 = 1e30f, d1 = 1e30f, d2 = 1e30f;
  int i0 = 0, i1 = 0, i2 = 0;
  for (int i = 0; i < nk; ++i) {
    float dx = K[i * 3 + 0] - ux, dy = K[i * 3 + 1] - uy, dz = K[i * 3 + 2] - uz;
    float d = dx * dx + dy * dy + dz * dz;
    if (d < d0)      { d2 = d1; i2 = i1; d1 = d0; i1 = i0; d0 = d; i0 = i; }
    else if (d < d1) { d2 = d1; i2 = i1; d1 = d; i1 = i; }
    else if (d < d2) { d2 = d;  i2 = i; }
  }
  float w0 = 1.f / (d0 + 1e-8f), w1 = 1.f / (d1 + 1e-8f), w2 = 1.f / (d2 + 1e-8f);
  float inv = 1.f / (w0 + w1 + w2);
  idx[g * 3 + 0] = i0; idx[g * 3 + 1] = i1; idx[g * 3 + 2] = i2;
  w[g * 3 + 0] = w0 * inv; w[g * 3 + 1] = w1 * inv; w[g * 3 + 2] = w2 * inv;
}

__global__ void build_fp2_kernel(const _Float16* __restrict__ l2p2h,
                                 const int* __restrict__ nn_idx,
                                 const float* __restrict__ nn_w,
                                 const float* __restrict__ l1p,
                                 _Float16* __restrict__ A, int total) {
  int e = blockIdx.x * blockDim.x + threadIdx.x;
  if (e >= total) return;
  int c = e % 576, bs = e / 576;
  int s = bs % 512, b = bs / 512;
  float v;
  if (c < 256) {
    int g = b * 512 + s;
    v = 0.f;
#pragma unroll
    for (int j = 0; j < 3; ++j) {
      int p = nn_idx[g * 3 + j];
      v += nn_w[g * 3 + j] * (float)l2p2h[((size_t)b * 128 + p) * 256 + c];
    }
  } else {
    v = l1p[((size_t)b * 320 + (c - 256)) * 512 + s];
  }
  A[e] = (_Float16)v;
}

__global__ void conv_last_sigmoid_kernel(const _Float16* __restrict__ A,
                                         LayerP L, float* __restrict__ xout,
                                         int total) {
  int g = blockIdx.x * blockDim.x + threadIdx.x;
  if (g >= total) return;
  float acc = L.b[0];
  const _Float16* r = A + (size_t)g * 128;
  for (int c = 0; c < 128; ++c) acc += (float)r[c] * L.W[c];
  xout[g] = 1.f / (1.f + expf(-acc));
}

__global__ void gf_kernel(const float* __restrict__ x,
                          const float* __restrict__ l1p,
                          float* __restrict__ obj, float* __restrict__ back,
                          int total) {
  int g = blockIdx.x * blockDim.x + threadIdx.x;
  if (g >= total) return;
  int b = g / 128, c = g % 128;
  float mo = -1e30f, mb = -1e30f;
  for (int s = 0; s < 512; ++s) {
    float xv = x[b * 512 + s];
    float f = l1p[((size_t)b * 128 + c) * 512 + s];
    mo = fmaxf(mo, xv * f);
    mb = fmaxf(mb, (1.f - xv) * f);
  }
  obj[g] = mo; back[g] = mb;
}

__global__ void final_seg_kernel(const float* __restrict__ xyz,
                                 const float* __restrict__ l1_xyz,
                                 const float* __restrict__ x,
                                 float* __restrict__ out, int total) {
  int g = blockIdx.x * blockDim.x + threadIdx.x;
  if (g >= total) return;
  int b = g / kN;
  float ux = xyz[g * 3 + 0], uy = xyz[g * 3 + 1], uz = xyz[g * 3 + 2];
  const float* K = l1_xyz + (size_t)b * 512 * 3;
  float d0 = 1e30f, d1 = 1e30f, d2 = 1e30f;
  int i0 = 0, i1 = 0, i2 = 0;
  for (int i = 0; i < 512; ++i) {
    float dx = K[i * 3 + 0] - ux, dy = K[i * 3 + 1] - uy, dz = K[i * 3 + 2] - uz;
    float d = dx * dx + dy * dy + dz * dz;
    if (d < d0)      { d2 = d1; i2 = i1; d1 = d0; i1 = i0; d0 = d; i0 = i; }
    else if (d < d1) { d2 = d1; i2 = i1; d1 = d; i1 = i; }
    else if (d < d2) { d2 = d;  i2 = i; }
  }
  out[g] = x[b * 512 + i0] + x[b * 512 + i1] + x[b * 512 + i2];
}

// ---------------------------------------------------------------------------
// Host-side orchestration
// ---------------------------------------------------------------------------
static inline LayerP takeLayer(void* const* d_in, int& i) {
  LayerP L;
  L.W     = (const float*)d_in[i++];
  L.b     = (const float*)d_in[i++];
  L.gamma = (const float*)d_in[i++];
  L.beta  = (const float*)d_in[i++];
  L.mean  = (const float*)d_in[i++];
  L.var   = (const float*)d_in[i++];
  return L;
}

extern "C" void kernel_launch(void* const* d_in, const int* in_sizes, int n_in,
                              void* d_out, int out_size, void* d_ws, size_t ws_size,
                              hipStream_t stream) {
  (void)in_sizes; (void)n_in; (void)out_size; (void)ws_size;
  int ii = 0;
  const float* xyz = (const float*)d_in[ii++];
  LayerP sa1[3][3], sa2[2][3], sa3[3], fp3[2], fp2[2], conv1[2];
  for (int br = 0; br < 3; ++br) for (int l = 0; l < 3; ++l) sa1[br][l] = takeLayer(d_in, ii);
  for (int br = 0; br < 2; ++br) for (int l = 0; l < 3; ++l) sa2[br][l] = takeLayer(d_in, ii);
  for (int l = 0; l < 3; ++l) sa3[l]   = takeLayer(d_in, ii);
  for (int l = 0; l < 2; ++l) fp3[l]   = takeLayer(d_in, ii);
  for (int l = 0; l < 2; ++l) fp2[l]   = takeLayer(d_in, ii);
  for (int l = 0; l < 2; ++l) conv1[l] = takeLayer(d_in, ii);

  // ---- workspace carve-out ----
  char* ws = (char*)d_ws;
  size_t off = 0;
  auto alloc = [&](size_t bytes) -> char* {
    char* p = ws + off;
    off += (bytes + 255) & ~(size_t)255;
    return p;
  };
  float* l1_xyz = (float*)alloc((size_t)kB * 512 * 3 * 4);
  float* l2_xyz = (float*)alloc((size_t)kB * 128 * 3 * 4);
  int*   fpsidx = (int*)  alloc((size_t)kB * 512 * 4);
  float* l1p    = (float*)alloc((size_t)kB * 320 * 512 * 4);   // sa1 out (B,320,512)
  float* l2p    = (float*)alloc((size_t)kB * 512 * 128 * 4);   // sa2 out (B,512,128)
  float* l3     = (float*)alloc((size_t)kB * 1024 * 4);        // sa3 out (B,1024)
  int*   idxbuf = (int*)  alloc((size_t)kB * 512 * 128 * 4);   // ball-query idx (reused)
  int*   nn_idx = (int*)  alloc((size_t)kB * 512 * 3 * 4);
  float* nn_w   = (float*)alloc((size_t)kB * 512 * 3 * 4);
  _Float16* A3h   = (_Float16*)alloc((size_t)2048 * 544 * 2);
  _Float16* H1    = (_Float16*)alloc((size_t)2048 * 256 * 2);
  _Float16* H2    = (_Float16*)alloc((size_t)2048 * 512 * 2);
  _Float16* H3    = (_Float16*)alloc((size_t)2048 * 1024 * 2);
  _Float16* Afp3  = (_Float16*)alloc((size_t)2048 * 1536 * 2);
  _Float16* Hfp3  = (_Float16*)alloc((size_t)2048 * 256 * 2);
  _Float16* l2p2h = (_Float16*)alloc((size_t)2048 * 256 * 2);
  _Float16* Afp2  = (_Float16*)alloc((size_t)8192 * 576 * 2);
  _Float16* Hfp2  = (_Float16*)alloc((size_t)8192 * 256 * 2);
  _Float16* l1ph  = (_Float16*)alloc((size_t)8192 * 128 * 2);
  _Float16* C1h   = (_Float16*)alloc((size_t)8192 * 128 * 2);

  // ---- pre-convert all GEMM weights to padded f16 ----
  auto wcvt = [&](const LayerP& L, int cout, int cin) -> _Float16* {
    int coutp = rup16(cout), kpad = rup32(cin);
    _Float16* Wh = (_Float16*)alloc((size_t)coutp * kpad * 2);
    int total = coutp * kpad;
    convert_w_kernel<<<(total + 255) / 256, 256, 0, stream>>>(L.W, cout, cin,
                                                              Wh, coutp, kpad);
    return Wh;
  };
  _Float16 *sa1w[3][3], *sa2w[2][3], *sa3w[3], *fp3w[2], *fp2w[2], *c1w;
  {
    const int sa1c[3][4] = {{6,32,32,64},{6,64,64,128},{6,64,96,128}};
    for (int br = 0; br < 3; ++br)
      for (int l = 0; l < 3; ++l)
        sa1w[br][l] = wcvt(sa1[br][l], sa1c[br][l+1], sa1c[br][l]);
    const int sa2c[2][4] = {{323,128,128,256},{323,128,196,256}};
    for (int br = 0; br < 2; ++br)
      for (int l = 0; l < 3; ++l)
        sa2w[br][l] = wcvt(sa2[br][l], sa2c[br][l+1], sa2c[br][l]);
    const int sa3c[4] = {515,256,512,1024};
    for (int l = 0; l < 3; ++l) sa3w[l] = wcvt(sa3[l], sa3c[l+1], sa3c[l]);
    const int fp3c[3] = {1536,256,256};
    for (int l = 0; l < 2; ++l) fp3w[l] = wcvt(fp3[l], fp3c[l+1], fp3c[l]);
    const int fp2c[3] = {576,256,128};
    for (int l = 0; l < 2; ++l) fp2w[l] = wcvt(fp2[l], fp2c[l+1], fp2c[l]);
    c1w = wcvt(conv1[0], 128, 128);
  }

  // ---- output carve-out ----
  float* out      = (float*)d_out;
  float* x_out    = out;                       // (B,1,512)
  float* l1p_out  = out + 8192;                // (B,128,512)
  float* obj_out  = l1p_out + (size_t)kB * 128 * 512;
  float* back_out = obj_out + kB * 128;
  float* seg_out  = back_out + kB * 128;       // (B,1,4096)

  // ================= SA1 (npoint=512, MSG) =================
  fps_kernel<kN, 512><<<kB, 256, 0, stream>>>(xyz, fpsidx, l1_xyz);

  // branch 0: r=0.1, K=32, [6,32,32,64], coff 0
  ball_query_kernel<<<32, 256, 0, stream>>>(xyz, kN, l1_xyz, kB * 512, 0.01f, 32, idxbuf);
  sa_group_mlp_kernel<32, 6, 32, 32, 32, 64, 32, 64>
      <<<kB * 512, 256, (32 * 32 + 32 * 64) * 2, stream>>>(
          xyz, kN, xyz, 3, 0, idxbuf, l1_xyz, 512,
          sa1w[0][0], sa1w[0][1], sa1w[0][2],
          sa1[0][0], sa1[0][1], sa1[0][2], l1p, 320, 0);
  // branch 1: r=0.2, K=64, [6,64,64,128], coff 64
  ball_query_kernel<<<32, 256, 0, stream>>>(xyz, kN, l1_xyz, kB * 512, 0.04f, 64, idxbuf);
  sa_group_mlp_kernel<64, 6, 32, 64, 64, 128, 64, 128>
      <<<kB * 512, 256, (64 * 64 + 64 * 128) * 2, stream>>>(
          xyz, kN, xyz, 3, 0, idxbuf, l1_xyz, 512,
          sa1w[1][0], sa1w[1][1], sa1w[1][2],
          sa1[1][0], sa1[1][1], sa1[1][2], l1p, 320, 64);
  // branch 2: r=0.4, K=128, [6,64,96,128], coff 192
  ball_query_kernel<<<32, 256, 0, stream>>>(xyz, kN, l1_xyz, kB * 512, 0.16f, 128, idxbuf);
  sa_group_mlp_kernel<128, 6, 32, 64, 96, 128, 96, 128>
      <<<kB * 512, 256, (128 * 96 + 128 * 128) * 2, stream>>>(
          xyz, kN, xyz, 3, 0, idxbuf, l1_xyz, 512,
          sa1w[2][0], sa1w[2][1], sa1w[2][2],
          sa1[2][0], sa1[2][1], sa1[2][2], l1p, 320, 192);

  // ================= SA2 (npoint=128, MSG) =================
  fps_kernel<512, 128><<<kB, 256, 0, stream>>>(l1_xyz, fpsidx, l2_xyz);

  // branch 0: r=0.4, K=64, [323,128,128,256], coff 0
  ball_query_kernel<<<8, 256, 0, stream>>>(l1_xyz, 512, l2_xyz, kB * 128, 0.16f, 64, idxbuf);
  sa_group_mlp_kernel<64, 323, 352, 128, 128, 256, 352, 256>
      <<<kB * 128, 256, (64 * 352 + 64 * 256) * 2, stream>>>(
          l1_xyz, 512, l1p, 320, 1, idxbuf, l2_xyz, 128,
          sa2w[0][0], sa2w[0][1], sa2w[0][2],
          sa2[0][0], sa2[0][1], sa2[0][2], l2p, 512, 0);
  // branch 1: r=0.8, K=128, [323,128,196,256], coff 256
  ball_query_kernel<<<8, 256, 0, stream>>>(l1_xyz, 512, l2_xyz, kB * 128, 0.64f, 128, idxbuf);
  sa_group_mlp_kernel<128, 323, 352, 128, 196, 256, 352, 256>
      <<<kB * 128, 256, (128 * 352 + 128 * 256) * 2, stream>>>(
          l1_xyz, 512, l1p, 320, 1, idxbuf, l2_xyz, 128,
          sa2w[1][0], sa2w[1][1], sa2w[1][2],
          sa2[1][0], sa2[1][1], sa2[1][2], l2p, 512, 256);

  // ================= SA3 (group-all, [515,256,512,1024]) =================
  build_sa3_kernel<<<(kB * 128 * 544 + 255) / 256, 256, 0, stream>>>(
      l2_xyz, l2p, A3h, kB * 128 * 544);
  gemm_bn_kernel<<<dim3(128, 16), 32, 0, stream>>>(A3h, 544, sa3w[0], 544, sa3[0], 256,
                                                   H1, 256, nullptr, 0, 1);
  gemm_bn_kernel<<<dim3(128, 32), 32, 0, stream>>>(H1, 256, sa3w[1], 256, sa3[1], 512,
                                                   H2, 512, nullptr, 0, 1);
  gemm_bn_kernel<<<dim3(128, 64), 32, 0, stream>>>(H2, 512, sa3w[2], 512, sa3[2], 1024,
                                                   H3, 1024, nullptr, 0, 1);
  maxpool_rows_kernel<<<(kB * 1024 + 255) / 256, 256, 0, stream>>>(H3, 128, 1024, l3,
                                                                   kB * 1024);

  // ================= FP3: [1536,256,256] (BN+ReLU both) =================
  build_fp3_kernel<<<(kB * 128 * 1536 + 255) / 256, 256, 0, stream>>>(
      l3, l2p, Afp3, kB * 128 * 1536);
  gemm_bn_kernel<<<dim3(128, 16), 32, 0, stream>>>(Afp3, 1536, fp3w[0], 1536, fp3[0], 256,
                                                   Hfp3, 256, nullptr, 0, 1);
  gemm_bn_kernel<<<dim3(128, 16), 32, 0, stream>>>(Hfp3, 256, fp3w[1], 256, fp3[1], 256,
                                                   l2p2h, 256, nullptr, 0, 1);

  // ================= three_nn + FP2: [576,256,128] =================
  three_nn_kernel<<<32, 256, 0, stream>>>(l1_xyz, 512, l2_xyz, 128, nn_idx, nn_w,
                                          kB * 512);
  build_fp2_kernel<<<(kB * 512 * 576 + 255) / 256, 256, 0, stream>>>(
      l2p2h, nn_idx, nn_w, l1p, Afp2, kB * 512 * 576);
  gemm_bn_kernel<<<dim3(512, 16), 32, 0, stream>>>(Afp2, 576, fp2w[0], 576, fp2[0], 256,
                                                   Hfp2, 256, nullptr, 0, 1);
  gemm_bn_kernel<<<dim3(512, 8), 32, 0, stream>>>(Hfp2, 256, fp2w[1], 256, fp2[1], 128,
                                                  l1ph, 128, l1p_out, 512, 1);

  // ================= conv1 + sigmoid =================
  gemm_bn_kernel<<<dim3(512, 8), 32, 0, stream>>>(l1ph, 128, c1w, 128, conv1[0], 128,
                                                  C1h, 128, nullptr, 0, 1);
  conv_last_sigmoid_kernel<<<32, 256, 0, stream>>>(C1h, conv1[1], x_out, kB * 512);

  // ================= global features + final seg =================
  gf_kernel<<<8, 256, 0, stream>>>(x_out, l1p_out, obj_out, back_out, kB * 128);
  final_seg_kernel<<<(kB * kN + 255) / 256, 256, 0, stream>>>(xyz, l1_xyz, x_out,
                                                              seg_out, kB * kN);
}
